// JoinConvNet_78391743087033
// MI455X (gfx1250) — matrix operand order repaired and checked by reading, not compiled, not run on hardware
//
#include <hip/hip_runtime.h>
#include <hip/hip_bf16.h>

// ---------------- CDNA5 WMMA types ----------------
typedef __bf16 bf16_t;
typedef __attribute__((ext_vector_type(16))) bf16_t       v16bf;
typedef __attribute__((ext_vector_type(8)))  float        v8f;
typedef __attribute__((ext_vector_type(4)))  unsigned int v4u;

// ---------------- problem sizes ----------------
constexpr int Bsz  = 1024;
constexpr int Lseq = 200;
constexpr int Emb  = 300;
constexpr int Cch  = 80;    // conv channels (GEMM N)
constexpr int Wwin = 3;     // conv window
constexpr int Osz  = 30;    // fc output

constexpr int Mpos  = Lseq - Wwin + 1;     // 198 valid positions (GEMM M)
constexpr int ES    = 64;                  // embed (K) slice staged per round
constexpr int NES   = (Emb + ES - 1) / ES; // 5 slices (last zero-padded)
constexpr int MT    = 13;                  // ceil(198/16) M tiles
constexpr int NTt   = 5;                   // 80/16 N tiles
constexpr int XROWS = MT * 16 + Wwin - 1;  // 210 staged x rows (padded + halo)

// round-to-nearest-even f32 -> bf16 (raw bits)
__device__ __forceinline__ unsigned int f2bf(float f) {
  unsigned int u = __float_as_uint(f);
  u += 0x7fffu + ((u >> 16) & 1u);
  return u >> 16;
}

// One workgroup = one (batch, subnet). Implicit-GEMM conv via bf16 WMMA,
// fused max-over-sequence + bias + ReLU epilogue. h: [B, C] fp32.
__global__ __launch_bounds__(256)
void conv_max_relu_kernel(const float* __restrict__ x,      // [B,1,L,E]
                          const float* __restrict__ wconv,  // [C,1,W,E]
                          const float* __restrict__ bconv,  // [C]
                          float* __restrict__ h)            // [B,C]
{
  __shared__ __align__(16) unsigned short xs[XROWS * ES];       // 26,880 B
  __shared__ __align__(16) unsigned short wt[Wwin * Cch * ES];  // 30,720 B
  __shared__ float red[8 * Cch];                                //  2,560 B

  const int b    = blockIdx.x;
  const int tid  = threadIdx.x;
  const int lane = tid & 31;   // wave32
  const int wave = tid >> 5;   // 8 waves
  const int nrow = lane & 15;  // row (A) / col (B) owned by this lane
  const int hl   = lane >> 4;  // lane half selects K sub-range

  // Every wave computes exactly 2 M tiles (compile-time trip counts so the
  // accumulators stay in VGPRs). Waves 5..7 duplicate tile 12; the max
  // epilogue makes duplicates harmless.
  const int mt0 = wave;
  const int mt1 = (wave + 8 < MT) ? (wave + 8) : (MT - 1);

  const v8f vzero = {0.f, 0.f, 0.f, 0.f, 0.f, 0.f, 0.f, 0.f};
  v8f acc[2][NTt];
  #pragma unroll
  for (int i = 0; i < 2; ++i)
    #pragma unroll
    for (int j = 0; j < NTt; ++j) acc[i][j] = vzero;

  const float* __restrict__ xb = x + (size_t)b * Lseq * Emb;

  for (int es = 0; es < NES; ++es) {
    const int e0 = es * ES;

    // ---- stage x rows [0..199] (+zero pad rows), float4 -> 4x bf16 ----
    for (int i = tid; i < XROWS * (ES / 4); i += 256) {
      const int row = i >> 4;
      const int q   = i & 15;
      const int e   = e0 + q * 4;
      float4 v = {0.f, 0.f, 0.f, 0.f};
      if (row < Lseq) {
        const float* p = &xb[row * Emb + e];
        if (e + 3 < Emb) {
          v = *(const float4*)p;
        } else {
          if (e + 0 < Emb) v.x = p[0];
          if (e + 1 < Emb) v.y = p[1];
          if (e + 2 < Emb) v.z = p[2];
        }
      }
      uint2 pk;
      pk.x = f2bf(v.x) | (f2bf(v.y) << 16);
      pk.y = f2bf(v.z) | (f2bf(v.w) << 16);
      *(uint2*)(&xs[row * ES + q * 4]) = pk;
    }
    // ---- stage transposed weight slice: wt[w][c][k] = W[c][w][e0+k] ----
    for (int i = tid; i < Wwin * Cch * (ES / 4); i += 256) {
      const int q = i & 15;            // 4-elem group within k
      const int c = (i >> 4) % Cch;
      const int w = i / (Cch * (ES / 4));
      const int e = e0 + q * 4;
      float4 v = {0.f, 0.f, 0.f, 0.f};
      const float* p = &wconv[(c * Wwin + w) * Emb + e];
      if (e + 3 < Emb) {
        v = *(const float4*)p;
      } else {
        if (e + 0 < Emb) v.x = p[0];
        if (e + 1 < Emb) v.y = p[1];
        if (e + 2 < Emb) v.z = p[2];
      }
      uint2 pk;
      pk.x = f2bf(v.x) | (f2bf(v.y) << 16);
      pk.y = f2bf(v.z) | (f2bf(v.w) << 16);
      *(uint2*)(&wt[(w * Cch + c) * ES + q * 4]) = pk;
    }
    __syncthreads();

    #pragma unroll
    for (int w = 0; w < Wwin; ++w) {
      #pragma unroll
      for (int k0 = 0; k0 < ES; k0 += 32) {
        // B fragments: contiguous per-lane runs -> two ds_load_b128 each
        v16bf bfr[NTt];
        #pragma unroll
        for (int nt = 0; nt < NTt; ++nt) {
          const int roff = (w * Cch + nt * 16 + nrow) * ES + k0 + hl * 16;
          union { v16bf v; v4u u[2]; } tb;
          tb.u[0] = *(const v4u*)(&wt[roff]);
          tb.u[1] = *(const v4u*)(&wt[roff + 8]);
          bfr[nt] = tb.v;
        }
        #pragma unroll
        for (int mi = 0; mi < 2; ++mi) {
          const int mt   = (mi == 0) ? mt0 : mt1;
          const int aoff = (mt * 16 + nrow + w) * ES + k0 + hl * 16;
          union { v16bf v; v4u u[2]; } ta;
          ta.u[0] = *(const v4u*)(&xs[aoff]);        // K = k0+hl*16+{0..7}
          ta.u[1] = *(const v4u*)(&xs[aoff + 8]);    // K = k0+hl*16+8+{0..7}
          #pragma unroll
          for (int nt = 0; nt < NTt; ++nt) {
            acc[mi][nt] = __builtin_amdgcn_wmma_f32_16x16x32_bf16(
                false, ta.v, false, bfr[nt], (short)0, acc[mi][nt],
                false, false);
          }
        }
      }
    }
    __syncthreads();
  }

  // ---- fused epilogue: masked max over positions, then +bias, ReLU ----
  #pragma unroll
  for (int nt = 0; nt < NTt; ++nt) {
    float m = -3.0e38f;
    #pragma unroll
    for (int mi = 0; mi < 2; ++mi) {
      const int mt = (mi == 0) ? mt0 : mt1;
      #pragma unroll
      for (int r = 0; r < 8; ++r) {
        const int p = mt * 16 + r + 8 * hl;  // C/D layout: M = r + 8*half
        const float v = acc[mi][nt][r];
        m = (p < Mpos) ? fmaxf(m, v) : m;
      }
    }
    m = fmaxf(m, __shfl_xor(m, 16, 32));  // combine lane halves (wave32)
    if (lane < 16) red[wave * Cch + nt * 16 + lane] = m;
  }
  __syncthreads();
  if (tid < Cch) {
    float m = -3.0e38f;
    #pragma unroll
    for (int wv = 0; wv < 8; ++wv) m = fmaxf(m, red[wv * Cch + tid]);
    h[b * Cch + tid] = fmaxf(m + bconv[tid], 0.f);
  }
}

// Tiny tail: o1 = relu(h1 @ wfc1^T + b1), o2 likewise, out[b] = dot(o1,o2).
__global__ __launch_bounds__(32)
void fc_dot_kernel(const float* __restrict__ h1, const float* __restrict__ h2,
                   const float* __restrict__ wfc1, const float* __restrict__ bfc1,
                   const float* __restrict__ wfc2, const float* __restrict__ bfc2,
                   float* __restrict__ out)
{
  const int b = blockIdx.x;
  const int i = threadIdx.x;
  float p = 0.f;
  if (i < Osz) {
    float s1 = bfc1[i], s2 = bfc2[i];
    #pragma unroll 4
    for (int c = 0; c < Cch; ++c) {
      s1 = fmaf(h1[b * Cch + c], wfc1[i * Cch + c], s1);
      s2 = fmaf(h2[b * Cch + c], wfc2[i * Cch + c], s2);
    }
    p = fmaxf(s1, 0.f) * fmaxf(s2, 0.f);
  }
  #pragma unroll
  for (int off = 16; off; off >>= 1) p += __shfl_xor(p, off, 32);
  if (i == 0) out[b] = p;
}

extern "C" void kernel_launch(void* const* d_in, const int* in_sizes, int n_in,
                              void* d_out, int out_size, void* d_ws, size_t ws_size,
                              hipStream_t stream) {
  (void)in_sizes; (void)n_in; (void)out_size; (void)ws_size;
  const float* x1     = (const float*)d_in[0];
  const float* x2     = (const float*)d_in[1];
  const float* wconv1 = (const float*)d_in[2];
  const float* bconv1 = (const float*)d_in[3];
  const float* wfc1   = (const float*)d_in[4];
  const float* bfc1   = (const float*)d_in[5];
  const float* wconv2 = (const float*)d_in[6];
  const float* bconv2 = (const float*)d_in[7];
  const float* wfc2   = (const float*)d_in[8];
  const float* bfc2   = (const float*)d_in[9];

  float* h1 = (float*)d_ws;                 // [B, C]
  float* h2 = h1 + (size_t)Bsz * Cch;       // [B, C]

  conv_max_relu_kernel<<<Bsz, 256, 0, stream>>>(x1, wconv1, bconv1, h1);
  conv_max_relu_kernel<<<Bsz, 256, 0, stream>>>(x2, wconv2, bconv2, h2);
  fc_dot_kernel<<<Bsz, 32, 0, stream>>>(h1, h2, wfc1, bfc1, wfc2, bfc2,
                                        (float*)d_out);
}
